// SwinTransformerBlock3D_28973849379145
// MI455X (gfx1250) — matrix operand order repaired and checked by reading, not compiled, not run on hardware
//
#include <hip/hip_runtime.h>
#include <hip/hip_bf16.h>
#include <cstdint>

// ---------------------------------------------------------------------------
// Types
// ---------------------------------------------------------------------------
typedef __attribute__((ext_vector_type(16))) _Float16 v16h;
typedef __attribute__((ext_vector_type(8)))  _Float16 h8;
typedef __attribute__((ext_vector_type(4)))  _Float16 h4;
typedef __attribute__((ext_vector_type(8)))  float    v8f;

union VH { v16h v; h8 h[2]; };

// ---------------------------------------------------------------------------
// Problem constants
// ---------------------------------------------------------------------------
#define Dd 16
#define Hh 112
#define Ww 112
#define Cc 128
#define NTOK 200704          // 16*112*112
#define NWIN 2048            // 8*16*16
#define NT 98                // tokens per window (2*7*7)
#define HEADS 4
#define HD 32
#define SCALE 0.17677669529663687f

// ---------------------------------------------------------------------------
// Helpers
// ---------------------------------------------------------------------------
__device__ __forceinline__ float wave_sum(float v) {
#pragma unroll
  for (int o = 16; o > 0; o >>= 1) v += __shfl_xor(v, o, 32);
  return v;
}
__device__ __forceinline__ float wave_max(float v) {
#pragma unroll
  for (int o = 16; o > 0; o >>= 1) v = fmaxf(v, __shfl_xor(v, o, 32));
  return v;
}

// LDS byte offset of a shared-memory pointer (AS3 pointers are 32-bit).
__device__ __forceinline__ uint32_t lds_off(const void* p) {
  return (uint32_t)(uintptr_t)(const __attribute__((address_space(3))) void*)p;
}

// CDNA5 async global->LDS copy, 16 bytes per lane (GVS addressing).
// VDST = per-lane LDS byte address, VADDR = per-lane 32-bit byte offset,
// SADDR = uniform 64-bit base. Tracked by ASYNCcnt.
__device__ __forceinline__ void async_copy16(uint32_t lds_addr, const void* gbase,
                                             uint32_t goff) {
  asm volatile("global_load_async_to_lds_b128 %0, %1, %2 offset:0"
               :: "v"(lds_addr), "v"(goff), "s"(gbase)
               : "memory");
}
__device__ __forceinline__ void wait_async_le6() {
  asm volatile("s_wait_asynccnt 0x6" ::: "memory");
}
__device__ __forceinline__ void wait_async_0() {
  asm volatile("s_wait_asynccnt 0x0" ::: "memory");
}

// window-token row index -> flat token index in ORIGINAL (unshifted) layout.
// win token at shifted coord g maps to original coord (g + shift) mod size,
// which serves both the LN1 gather (roll by -shift) and the proj scatter
// (window_reverse + roll by +shift).
__device__ __forceinline__ long win_to_orig(int ridx) {
  int w = ridx / NT, t = ridx % NT;
  int wd = w >> 8, wh = (w >> 4) & 15, ww = w & 15;
  int ld = t / 49, r = t % 49;
  int lh = r / 7, lw = r % 7;
  int gd = wd * 2 + ld, gh = wh * 7 + lh, gw = ww * 7 + lw;
  int od = (gd + 1) & 15;
  int oh = gh + 3; if (oh >= Hh) oh -= Hh;
  int ow = gw + 3; if (ow >= Ww) ow -= Ww;
  return (long)(od * Hh + oh) * Ww + ow;
}

// ---------------------------------------------------------------------------
// f32 -> f16 weight convert
// ---------------------------------------------------------------------------
__global__ void f32_to_f16(const float* __restrict__ in, _Float16* __restrict__ out, int n) {
  int i = blockIdx.x * 256 + threadIdx.x;
  if (i < n) out[i] = (_Float16)in[i];
}

// ---------------------------------------------------------------------------
// LayerNorm (one wave per token), optional shift+window-partition gather.
// ---------------------------------------------------------------------------
template <int MAPPED>
__global__ __launch_bounds__(256) void ln_kernel(const float* __restrict__ xin,
                                                 const float* __restrict__ w,
                                                 const float* __restrict__ b,
                                                 _Float16* __restrict__ out) {
  int wave = threadIdx.x >> 5, lane = threadIdx.x & 31;
  long ridx = (long)blockIdx.x * 8 + wave;
  long src = MAPPED ? win_to_orig((int)ridx) * (long)Cc : ridx * Cc;
  float4 xv = *(const float4*)&xin[src + lane * 4];
  float s  = xv.x + xv.y + xv.z + xv.w;
  float s2 = xv.x * xv.x + xv.y * xv.y + xv.z * xv.z + xv.w * xv.w;
  s = wave_sum(s);
  s2 = wave_sum(s2);
  float mean = s * (1.0f / Cc);
  float var  = s2 * (1.0f / Cc) - mean * mean;
  float rstd = rsqrtf(var + 1e-5f);
  float4 wv = *(const float4*)&w[lane * 4];
  float4 bv = *(const float4*)&b[lane * 4];
  h4 o;
  o[0] = (_Float16)((xv.x - mean) * rstd * wv.x + bv.x);
  o[1] = (_Float16)((xv.y - mean) * rstd * wv.y + bv.y);
  o[2] = (_Float16)((xv.z - mean) * rstd * wv.z + bv.z);
  o[3] = (_Float16)((xv.w - mean) * rstd * wv.w + bv.w);
  *(h4*)&out[ridx * Cc + lane * 4] = o;
}

// ---------------------------------------------------------------------------
// WMMA GEMM: out[M,N] = A[M,K](f16) @ W[N,K]^T(f16) + bias, with epilogues.
// K, N compile-time: fully unrolled double-buffered async pipeline.
// Block tile 64x128, 8 waves x (32x32); 6 async b128 copies/thread/K-tile.
// EPI 0: f16 out + bias                        (QKV)
// EPI 1: f16 out + bias + exact GELU           (FC1)
// EPI 2: f32 scatter: x_res[map(m)] = res[map(m)] + v   (proj+reverse+roll+res)
// EPI 3: f32 out: d_out[m] = v + res[m]        (FC2 + residual)
// ---------------------------------------------------------------------------
template <int EPI, int K, int N>
__global__ __launch_bounds__(256) void gemm_wmma(const _Float16* __restrict__ A,
                                                 const _Float16* __restrict__ W,
                                                 const float* __restrict__ bias,
                                                 void* __restrict__ outp,
                                                 const float* __restrict__ res) {
  __shared__ __align__(16) _Float16 As[2][64 * 64];
  __shared__ __align__(16) _Float16 Bs[2][128 * 64];

  int tid = threadIdx.x;
  int wave = tid >> 5, lane = tid & 31;
  int wm = wave >> 2, wn = wave & 3;
  int lrow = lane & 15;
  bool hib = lane >= 16;
  long row0 = (long)blockIdx.x * 64;
  int col0 = blockIdx.y * 128;

  uint32_t as_base = lds_off(&As[0][0]);
  uint32_t bs_base = lds_off(&Bs[0][0]);

  auto issue_tile = [&](int k0, int buf) {
#pragma unroll
    for (int p = 0; p < 2; p++) {       // A tile: 64x64 halves
      int idx = p * 256 + tid;
      int r = idx >> 3, c8 = (idx & 7) * 8;
      uint32_t lds = as_base + (uint32_t)(buf * (64 * 64 * 2) + (r * 64 + c8) * 2);
      uint32_t go = (uint32_t)(((size_t)(row0 + r) * K + k0 + c8) * 2);
      async_copy16(lds, A, go);
    }
#pragma unroll
    for (int p = 0; p < 4; p++) {       // B tile: 128x64 halves
      int idx = p * 256 + tid;
      int r = idx >> 3, c8 = (idx & 7) * 8;
      uint32_t lds = bs_base + (uint32_t)(buf * (128 * 64 * 2) + (r * 64 + c8) * 2);
      uint32_t go = (uint32_t)(((size_t)(col0 + r) * K + k0 + c8) * 2);
      async_copy16(lds, W, go);
    }
  };

  v8f acc[2][2] = {};

  constexpr int nk = K >> 6;
  issue_tile(0, 0);
#pragma unroll
  for (int i = 0; i < nk; i++) {
    constexpr int one = 1;
    int buf = i & one;
    if (i + 1 < nk) {
      issue_tile((i + 1) << 6, buf ^ 1);
      wait_async_le6();                 // tile i complete (in-order), tile i+1 in flight
    } else {
      wait_async_0();
    }
    __syncthreads();

#pragma unroll
    for (int kk = 0; kk < 64; kk += 32) {
      VH a[2], b[2];
#pragma unroll
      for (int t = 0; t < 2; t++) {
        int m = wm * 32 + t * 16 + lrow;
        int kb = kk + (hib ? 8 : 0);
        a[t].h[0] = *(const h8*)&As[buf][m * 64 + kb];
        a[t].h[1] = *(const h8*)&As[buf][m * 64 + kb + 16];
        int n = wn * 32 + t * 16 + lrow;
        int kb2 = kk + (hib ? 8 : 0);
        b[t].h[0] = *(const h8*)&Bs[buf][n * 64 + kb2];
        b[t].h[1] = *(const h8*)&Bs[buf][n * 64 + kb2 + 16];
      }
#pragma unroll
      for (int tm = 0; tm < 2; tm++)
#pragma unroll
        for (int tn = 0; tn < 2; tn++)
          acc[tm][tn] = __builtin_amdgcn_wmma_f32_16x16x32_f16(
              false, a[tm].v, false, b[tn].v, (short)0, acc[tm][tn], false, false);
    }
    __syncthreads();  // all waves done reading buf before it is refilled
  }

  // epilogue
#pragma unroll
  for (int tm = 0; tm < 2; tm++) {
#pragma unroll
    for (int tn = 0; tn < 2; tn++) {
      int n = col0 + wn * 32 + tn * 16 + lrow;
      float bv = bias[n];
#pragma unroll
      for (int r = 0; r < 8; r++) {
        long m = row0 + wm * 32 + tm * 16 + r + (hib ? 8 : 0);
        float v = acc[tm][tn][r] + bv;
        if (EPI == 0) {
          ((_Float16*)outp)[m * N + n] = (_Float16)v;
        } else if (EPI == 1) {
          float g = 0.5f * v * (1.0f + erff(v * 0.70710678118654752f));
          ((_Float16*)outp)[m * N + n] = (_Float16)g;
        } else if (EPI == 2) {
          long o = win_to_orig((int)m) * Cc + n;
          ((float*)outp)[o] = res[o] + v;
        } else {
          ((float*)outp)[m * N + n] = v + res[m * N + n];
        }
      }
    }
  }
}

// ---------------------------------------------------------------------------
// Fused windowed attention: one block per (window, head).
// S = scale*(q k^T) + rel_pos_bias + shift_mask ; softmax ; O = P V
// ---------------------------------------------------------------------------
__global__ __launch_bounds__(256) void attn_kernel(const _Float16* __restrict__ qkv,
                                                   const float* __restrict__ rpb,
                                                   _Float16* __restrict__ out) {
  __shared__ __align__(16) _Float16 Qs[112 * 32];
  __shared__ __align__(16) _Float16 Ks[112 * 32];
  __shared__ __align__(16) _Float16 Vt[32 * 128];   // transposed: [hd][token]
  __shared__ __align__(16) _Float16 SP[112 * 128];  // logits then probs (f16)
  __shared__ int   tinfo[112];                      // label<<9 | ld<<6 | lh<<3 | lw
  __shared__ float rbias[507];                      // this head's rel-pos-bias column

  int win = blockIdx.x >> 2;
  int head = blockIdx.x & 3;
  int tid = threadIdx.x;
  int wave = tid >> 5, lane = tid & 31;
  int lrow = lane & 15;
  bool hib = lane >= 16;

  uint32_t qs_base = lds_off(&Qs[0]);
  uint32_t ks_base = lds_off(&Ks[0]);

  // --- stage q,k via async global->LDS (contiguous 16B chunks) ---
  for (int idx = tid; idx < NT * 4; idx += 256) {
    int t = idx >> 2, c8 = (idx & 3) * 8;
    size_t base = (size_t)(win * NT + t) * 384 + head * 32 + c8;
    uint32_t l = (uint32_t)((t * 32 + c8) * 2);
    async_copy16(qs_base + l, qkv, (uint32_t)(base * 2));
    async_copy16(ks_base + l, qkv, (uint32_t)((base + 128) * 2));
  }
  // zero pad q/k rows 98..111
  if (tid < 56) {
    int t = NT + (tid >> 2), c8 = (tid & 3) * 8;
    h8 z = {};
    *(h8*)&Qs[t * 32 + c8] = z;
    *(h8*)&Ks[t * 32 + c8] = z;
  }
  // --- stage v transposed (VALU path: scattered LDS stores) ---
  for (int idx = tid; idx < 512; idx += 256) {
    int t = idx >> 2, c8 = (idx & 3) * 8;
    h8 v = {};
    if (t < NT) v = *(const h8*)&qkv[(size_t)(win * NT + t) * 384 + 256 + head * 32 + c8];
#pragma unroll
    for (int u = 0; u < 8; u++) Vt[(c8 + u) * 128 + t] = v[u];
  }
  // --- per-token packed coords + shift-mask region label ---
  if (tid < 112) {
    int t = tid;
    int info = 0;
    if (t < NT) {
      int wd = win >> 8, wh = (win >> 4) & 15, ww = win & 15;
      int ld = t / 49, r = t % 49;
      int lh = r / 7, lw = r % 7;
      int gd = wd * 2 + ld, gh = wh * 7 + lh, gw = ww * 7 + lw;
      int cd = (gd < 14) ? 0 : ((gd < 15) ? 1 : 2);
      int ch = (gh < 105) ? 0 : ((gh < 109) ? 1 : 2);
      int cw = (gw < 105) ? 0 : ((gw < 109) ? 1 : 2);
      int lab = (cd * 3 + ch) * 3 + cw;
      info = (lab << 9) | (ld << 6) | (lh << 3) | lw;
    }
    tinfo[t] = info;
  }
  // --- stage this head's bias column into LDS ---
  for (int idx = tid; idx < 507; idx += 256) rbias[idx] = rpb[idx * HEADS + head];

  wait_async_0();
  __syncthreads();

  // --- S = q k^T : 7x7 tiles of 16x16, K=32 (single WMMA each) ---
  for (int tile = wave; tile < 49; tile += 8) {
    int ti = tile / 7, tj = tile % 7;
    VH a, b;
    {
      int m = ti * 16 + lrow;
      int kb = hib ? 8 : 0;
      a.h[0] = *(const h8*)&Qs[m * 32 + kb];
      a.h[1] = *(const h8*)&Qs[m * 32 + kb + 16];
    }
    {
      int n = tj * 16 + lrow;
      int kb2 = hib ? 8 : 0;
      b.h[0] = *(const h8*)&Ks[n * 32 + kb2];
      b.h[1] = *(const h8*)&Ks[n * 32 + kb2 + 16];
    }
    v8f acc = {};
    acc = __builtin_amdgcn_wmma_f32_16x16x32_f16(false, a.v, false, b.v, (short)0,
                                                 acc, false, false);
    int j = tj * 16 + lrow;
    int cj = tinfo[j];
    int jd = (cj >> 6) & 1, jh = (cj >> 3) & 7, jw = cj & 7, jl = cj >> 9;
#pragma unroll
    for (int r = 0; r < 8; r++) {
      int i = ti * 16 + r + (hib ? 8 : 0);
      float s = -30000.0f;
      if (i < NT && j < NT) {
        int ci = tinfo[i];
        int idx = (((ci >> 6) & 1) - jd + 1) * 169 +
                  (((ci >> 3) & 7) - jh + 6) * 13 + ((ci & 7) - jw + 6);
        float msk = ((ci >> 9) == jl) ? 0.0f : -100.0f;
        s = acc[r] * SCALE + rbias[idx] + msk;
      }
      SP[i * 128 + j] = (_Float16)s;
    }
  }
  __syncthreads();

  // --- softmax rows (in place, f16 logits -> f16 probs) ---
  for (int i = wave; i < 112; i += 8) {
    if (i >= NT) {
      for (int j = lane; j < 128; j += 32) SP[i * 128 + j] = (_Float16)0.0f;
    } else {
      float mx = -1e30f;
      for (int j = lane; j < NT; j += 32) mx = fmaxf(mx, (float)SP[i * 128 + j]);
      mx = wave_max(mx);
      float sm = 0.0f;
      for (int j = lane; j < NT; j += 32) sm += __expf((float)SP[i * 128 + j] - mx);
      sm = wave_sum(sm);
      float inv = 1.0f / sm;
      for (int j = lane; j < 128; j += 32) {
        float p = (j < NT) ? __expf((float)SP[i * 128 + j] - mx) * inv : 0.0f;
        SP[i * 128 + j] = (_Float16)p;
      }
    }
  }
  __syncthreads();

  // --- O = P V : 7x2 tiles, K = 128 (padded tokens contribute zero) ---
  for (int tile = wave; tile < 14; tile += 8) {
    int ti = tile >> 1, tn = tile & 1;
    v8f acc = {};
#pragma unroll
    for (int kk = 0; kk < 128; kk += 32) {
      VH a, b;
      int m = ti * 16 + lrow;
      int kb = kk + (hib ? 8 : 0);
      a.h[0] = *(const h8*)&SP[m * 128 + kb];
      a.h[1] = *(const h8*)&SP[m * 128 + kb + 16];
      int n = tn * 16 + lrow;
      int kb2 = kk + (hib ? 8 : 0);
      b.h[0] = *(const h8*)&Vt[n * 128 + kb2];
      b.h[1] = *(const h8*)&Vt[n * 128 + kb2 + 16];
      acc = __builtin_amdgcn_wmma_f32_16x16x32_f16(false, a.v, false, b.v, (short)0,
                                                   acc, false, false);
    }
    int col = tn * 16 + lrow;
#pragma unroll
    for (int r = 0; r < 8; r++) {
      int i = ti * 16 + r + (hib ? 8 : 0);
      if (i < NT)
        out[(size_t)(win * NT + i) * Cc + head * HD + col] = (_Float16)acc[r];
    }
  }
}

// ---------------------------------------------------------------------------
// Launch
// ---------------------------------------------------------------------------
extern "C" void kernel_launch(void* const* d_in, const int* in_sizes, int n_in,
                              void* d_out, int out_size, void* d_ws, size_t ws_size,
                              hipStream_t stream) {
  const float* x    = (const float*)d_in[0];
  const float* n1w  = (const float*)d_in[1];
  const float* n1b  = (const float*)d_in[2];
  const float* qkvw = (const float*)d_in[3];
  const float* qkvb = (const float*)d_in[4];
  const float* rpb  = (const float*)d_in[5];
  const float* pjw  = (const float*)d_in[6];
  const float* pjb  = (const float*)d_in[7];
  const float* n2w  = (const float*)d_in[8];
  const float* n2b  = (const float*)d_in[9];
  const float* f1w  = (const float*)d_in[10];
  const float* f1b  = (const float*)d_in[11];
  const float* f2w  = (const float*)d_in[12];
  const float* f2b  = (const float*)d_in[13];

  char* ws = (char*)d_ws;
  // f16 weights
  _Float16* wq = (_Float16*)(ws + 0);        // 384*128
  _Float16* wp = (_Float16*)(ws + 98304);    // 128*128
  _Float16* w1 = (_Float16*)(ws + 131072);   // 512*128
  _Float16* w2 = (_Float16*)(ws + 262144);   // 128*512
  // activations (with reuse)
  const size_t OFF_WIN  = 524288;
  const size_t OFF_QKV  = OFF_WIN + (size_t)NTOK * Cc * 2;        // +51,380,224
  const size_t OFF_ATT  = OFF_QKV + (size_t)NTOK * 384 * 2;       // +154,140,672
  const size_t OFF_XRES = OFF_ATT + (size_t)NTOK * Cc * 2 + 1024; // +51,381,248
  _Float16* win  = (_Float16*)(ws + OFF_WIN);
  _Float16* qkv  = (_Float16*)(ws + OFF_QKV);
  _Float16* attn = (_Float16*)(ws + OFF_ATT);
  float*    xres = (float*)(ws + OFF_XRES);
  _Float16* h2   = win;                       // reuse (win consumed by QKV gemm)
  _Float16* fc1o = qkv;                       // reuse qkv+attn regions (205,521,920 B)

  // 1) weight conversion
  f32_to_f16<<<192, 256, 0, stream>>>(qkvw, wq, 49152);
  f32_to_f16<<<64,  256, 0, stream>>>(pjw,  wp, 16384);
  f32_to_f16<<<256, 256, 0, stream>>>(f1w,  w1, 65536);
  f32_to_f16<<<256, 256, 0, stream>>>(f2w,  w2, 65536);

  // 2) LN1 + cyclic shift + window partition -> f16
  ln_kernel<1><<<NTOK / 8, 256, 0, stream>>>(x, n1w, n1b, win);

  // 3) QKV: [200704,128] @ [384,128]^T
  gemm_wmma<0, 128, 384><<<dim3(NTOK / 64, 3), 256, 0, stream>>>(win, wq, qkvb, qkv,
                                                                 nullptr);

  // 4) fused windowed attention (bias + mask + softmax + PV)
  attn_kernel<<<NWIN * HEADS, 256, 0, stream>>>(qkv, rpb, attn);

  // 5) proj + window reverse + roll + residual -> x_res (f32)
  gemm_wmma<2, 128, 128><<<dim3(NTOK / 64, 1), 256, 0, stream>>>(attn, wp, pjb, xres, x);

  // 6) LN2
  ln_kernel<0><<<NTOK / 8, 256, 0, stream>>>(xres, n2w, n2b, h2);

  // 7) FC1 + exact GELU -> f16
  gemm_wmma<1, 128, 512><<<dim3(NTOK / 64, 4), 256, 0, stream>>>(h2, w1, f1b, fc1o,
                                                                 nullptr);

  // 8) FC2 + residual -> d_out (f32)
  gemm_wmma<3, 512, 128><<<dim3(NTOK / 64, 1), 256, 0, stream>>>(fc1o, w2, f2b, d_out,
                                                                 xres);
}
